// CausalSelfAttention_10479720203079
// MI455X (gfx1250) — matrix operand order repaired and checked by reading, not compiled, not run on hardware
//
#include <hip/hip_runtime.h>

#define DEV __device__ __forceinline__

typedef __attribute__((ext_vector_type(16))) __bf16 v16bf;
typedef __attribute__((ext_vector_type(8)))  float  v8f;

static constexpr int Bsz   = 8;
static constexpr int T     = 1024;
static constexpr int C     = 768;
static constexpr int H     = 12;
static constexpr int Dh    = 64;
static constexpr int M_TOT = Bsz * T;   // 8192
static constexpr int N_QKV = 3 * C;     // 2304

// ---------- bf16 helpers (raw-bit representation in unsigned short) ----------
DEV unsigned short f2bfbits(float f) {
    unsigned u = __float_as_uint(f);
    u += 0x7FFFu + ((u >> 16) & 1u);   // round-to-nearest-even
    return (unsigned short)(u >> 16);
}
DEV __bf16 bits2bf(unsigned short s) {
    union { unsigned short u; __bf16 b; } c; c.u = s; return c.b;
}

// A-operand K offset for v_wmma_*_16x16x32_bf16 (pair v = 0..7, hi = lane>>4)
DEV int kmapA(int v, int hi) { return ((v >= 4) ? 16 : 0) + hi * 8 + (v & 3) * 2; }
// B-operand K offset (same K order as A: lanes 0-15 K=0..7,16..23; lanes 16-31 K=8..15,24..31)
DEV int kmapB(int v, int hi) { return ((v >= 4) ? 16 : 0) + hi * 8 + (v & 3) * 2; }

// load one aligned bf16 pair (2 bytes each) into vector elements j, j+1
DEV void ld_pair(const unsigned short* p, v16bf& d, int j) {
    unsigned w = *(const unsigned*)p;
    d[j]     = bits2bf((unsigned short)(w & 0xFFFFu));
    d[j + 1] = bits2bf((unsigned short)(w >> 16));
}

// ---------------------------- prep kernels ----------------------------------
__global__ void k_cvt_bf16(const float* __restrict__ in,
                           unsigned short* __restrict__ out, int n) {
    int i = blockIdx.x * blockDim.x + threadIdx.x;
    if (i < n) out[i] = f2bfbits(in[i]);
}

// in: [K][N] fp32 row-major -> out: [N][K] bf16
__global__ void k_transpose_bf16(const float* __restrict__ in,
                                 unsigned short* __restrict__ out, int K, int N) {
    int i = blockIdx.x * blockDim.x + threadIdx.x;
    if (i < K * N) {
        int k = i / N, n = i % N;
        out[(size_t)n * K + k] = f2bfbits(in[i]);
    }
}

// ------------------- QKV GEMM + bias + RoPE, bf16 outputs -------------------
// Grid: (M_TOT/64, N_QKV/64), block 128 (4 waves); wave owns 16 M rows x 64 N.
// Q,K stored [B,H,T,D] (rope'd); V stored TRANSPOSED [B,H,D,T] so the PV
// B-fragment loads in the flash kernel are contiguous along the WMMA K-dim.
__global__ void __launch_bounds__(128) k_qkv_rope(
    const unsigned short* __restrict__ xb,    // [8192][768] bf16
    const unsigned short* __restrict__ WaT,   // [2304][768] bf16 (W_attn^T)
    const float* __restrict__ b_attn,         // [2304]
    unsigned short* __restrict__ Qb,          // [96][1024][64] bf16
    unsigned short* __restrict__ Kb,          // [96][1024][64] bf16
    unsigned short* __restrict__ Vt)          // [96][64][1024] bf16 (transposed)
{
    const int lane = threadIdx.x & 31;
    const int wave = threadIdx.x >> 5;
    const int hi = lane >> 4, ln = lane & 15;
    const int m0 = blockIdx.x * 64 + wave * 16;
    const int n0 = blockIdx.y * 64;

    v8f acc[4];
#pragma unroll
    for (int nt = 0; nt < 4; ++nt)
#pragma unroll
        for (int i = 0; i < 8; ++i) acc[nt][i] = 0.f;

    const unsigned short* arow = xb + (size_t)(m0 + ln) * C;
    for (int k0 = 0; k0 < C; k0 += 32) {
        __builtin_prefetch(arow + k0 + 32, 0, 1);
        v16bf a;
#pragma unroll
        for (int v = 0; v < 8; ++v) ld_pair(arow + k0 + kmapA(v, hi), a, v * 2);
#pragma unroll
        for (int nt = 0; nt < 4; ++nt) {
            const unsigned short* brow = WaT + (size_t)(n0 + nt * 16 + ln) * C + k0;
            v16bf b;
#pragma unroll
            for (int v = 0; v < 8; ++v) ld_pair(brow + kmapB(v, hi), b, v * 2);
            acc[nt] = __builtin_amdgcn_wmma_f32_16x16x32_bf16(
                false, a, false, b, (short)0, acc[nt], false, false);
        }
    }

    // Epilogue: bias + RoPE (q,k) and scatter.
    const int m_base = m0 + hi * 8;            // rows m_base..m_base+7 (C/D layout)
    const int b_ = m_base >> 10;
    const int t0 = m_base & 1023;              // t = t0 + i, contiguous
#pragma unroll
    for (int nt = 0; nt < 4; ++nt) {
        const int n = n0 + nt * 16 + ln;   // global qkv column
        const int which = n / C;           // 0=q 1=k 2=v (uniform per wave)
        const int c = n % C;
        const int h = c >> 6, d = c & 63;
        const float bias = b_attn[n];
        if (which == 2) {
            // pack 8 consecutive-t bf16 values -> one 16B store into Vt[bh][d][t0..t0+7]
            unsigned short vb16[8];
#pragma unroll
            for (int i = 0; i < 8; ++i) vb16[i] = f2bfbits(acc[nt][i] + bias);
            uint4 w;
            w.x = (unsigned)vb16[0] | ((unsigned)vb16[1] << 16);
            w.y = (unsigned)vb16[2] | ((unsigned)vb16[3] << 16);
            w.z = (unsigned)vb16[4] | ((unsigned)vb16[5] << 16);
            w.w = (unsigned)vb16[6] | ((unsigned)vb16[7] << 16);
            *(uint4*)(Vt + ((size_t)(b_ * H + h) * Dh + d) * T + t0) = w;
        } else {
            const float freq =
                __expf(-(float)(d & ~1) * (9.2103403719761836f / 64.f));
#pragma unroll
            for (int i = 0; i < 8; ++i) {
                const int t = t0 + i;
                float v = acc[nt][i] + bias;
                float partner = __shfl_xor(v, 1, 32);   // paired d (adjacent lane)
                float ang = (float)t * freq;
                float cw = __cosf(ang), sw = __sinf(ang);
                float r = (d & 1) ? (v * cw + partner * sw)
                                  : (v * cw - partner * sw);
                const size_t idx = ((size_t)(b_ * H + h) * T + t) * Dh + d;
                if (which == 0) Qb[idx] = f2bfbits(r);
                else            Kb[idx] = f2bfbits(r);
            }
        }
    }
}

// -------------------- fused causal flash attention --------------------------
// Grid: (T/64, B*H), block 128 (4 waves). Wave owns 16 q rows.
__global__ void __launch_bounds__(128) k_flash(
    const unsigned short* __restrict__ Qb,    // [96][1024][64]
    const unsigned short* __restrict__ Kb,    // [96][1024][64]
    const unsigned short* __restrict__ Vt,    // [96][64][1024]
    unsigned short* __restrict__ Yb)          // [8192][768] bf16 ([B,T,H,D])
{
    __shared__ unsigned short pt[4][16 * 32];  // per-wave P tile (16 q x 32 k)

    const int lane = threadIdx.x & 31;
    const int wave = threadIdx.x >> 5;
    const int hi = lane >> 4, ln = lane & 15;
    const int bh = blockIdx.y;                 // 0..95
    const int q0 = blockIdx.x * 64 + wave * 16;
    const size_t base = (size_t)bh * T * Dh;   // same total size for Vt

    // Q fragments (two 32-wide d slabs), A layout
    v16bf qa0, qa1;
    {
        const unsigned short* qrow = Qb + base + (size_t)(q0 + ln) * Dh;
#pragma unroll
        for (int v = 0; v < 8; ++v) {
            ld_pair(qrow + kmapA(v, hi), qa0, v * 2);
            ld_pair(qrow + 32 + kmapA(v, hi), qa1, v * 2);
        }
    }

    float rmax[8], rsum[8];
#pragma unroll
    for (int i = 0; i < 8; ++i) { rmax[i] = -3.0e38f; rsum[i] = 0.f; }
    v8f oacc[4];
#pragma unroll
    for (int nt = 0; nt < 4; ++nt)
#pragma unroll
        for (int i = 0; i < 8; ++i) oacc[nt][i] = 0.f;

    unsigned short* ptw = pt[wave];

    for (int k0 = 0; k0 < q0 + 16; k0 += 32) {
        // prefetch next tile's K row and V row for this lane
        __builtin_prefetch(Kb + base + (size_t)(k0 + 32 + ln) * Dh, 0, 1);
        __builtin_prefetch(Vt + base + (size_t)ln * T + k0 + 32, 0, 1);

        // S = Q K^T for two 16-column subtiles (chain two K=32 WMMAs over d)
        v8f s[2];
#pragma unroll
        for (int st = 0; st < 2; ++st) {
            const unsigned short* krow =
                Kb + base + (size_t)(k0 + st * 16 + ln) * Dh;
            v16bf kb0, kb1;
#pragma unroll
            for (int v = 0; v < 8; ++v) {
                ld_pair(krow + kmapB(v, hi), kb0, v * 2);
                ld_pair(krow + 32 + kmapB(v, hi), kb1, v * 2);
            }
            v8f z;
#pragma unroll
            for (int i = 0; i < 8; ++i) z[i] = 0.f;
            z = __builtin_amdgcn_wmma_f32_16x16x32_bf16(
                false, qa0, false, kb0, (short)0, z, false, false);
            z = __builtin_amdgcn_wmma_f32_16x16x32_bf16(
                false, qa1, false, kb1, (short)0, z, false, false);
            s[st] = z;
        }

        // online softmax over these 32 columns; rows m = i + 8*hi
        const int kc0 = k0 + ln, kc1 = k0 + 16 + ln;
        float alpha[8];
#pragma unroll
        for (int i = 0; i < 8; ++i) {
            const int q = q0 + hi * 8 + i;
            float s0 = s[0][i] * 0.125f; if (kc0 > q) s0 = -3.0e38f;
            float s1 = s[1][i] * 0.125f; if (kc1 > q) s1 = -3.0e38f;
            float mx = fmaxf(s0, s1);
#pragma unroll
            for (int msk = 1; msk < 16; msk <<= 1)
                mx = fmaxf(mx, __shfl_xor(mx, msk, 32));
            float mnew = fmaxf(rmax[i], mx);
            float a = __expf(rmax[i] - mnew);
            rmax[i] = mnew;
            float p0 = __expf(s0 - mnew);
            float p1 = __expf(s1 - mnew);
            float ps = p0 + p1;
#pragma unroll
            for (int msk = 1; msk < 16; msk <<= 1)
                ps += __shfl_xor(ps, msk, 32);
            rsum[i] = rsum[i] * a + ps;
            alpha[i] = a;
            // P (bf16) to LDS, D layout -> row m, 32 cols
            ptw[(hi * 8 + i) * 32 + ln]      = f2bfbits(p0);
            ptw[(hi * 8 + i) * 32 + 16 + ln] = f2bfbits(p1);
        }

        // rescale running O by alpha (same row mapping as C/D layout)
#pragma unroll
        for (int nt = 0; nt < 4; ++nt)
#pragma unroll
            for (int i = 0; i < 8; ++i) oacc[nt][i] *= alpha[i];

        __builtin_amdgcn_wave_barrier();

        // P as A fragment from LDS (cross-lane transpose)
        v16bf pa;
        {
            const unsigned short* prow = ptw + ln * 32;
#pragma unroll
            for (int v = 0; v < 8; ++v) {
                unsigned w = *(const unsigned*)(prow + kmapA(v, hi));
                pa[v * 2]     = bits2bf((unsigned short)(w & 0xFFFFu));
                pa[v * 2 + 1] = bits2bf((unsigned short)(w >> 16));
            }
        }
        // O += P * V : B fragment direct from transposed V, contiguous along k
#pragma unroll
        for (int nt = 0; nt < 4; ++nt) {
            const unsigned short* vrow =
                Vt + base + (size_t)(nt * 16 + ln) * T + k0;
            v16bf vb;
#pragma unroll
            for (int v = 0; v < 8; ++v) ld_pair(vrow + kmapB(v, hi), vb, v * 2);
            oacc[nt] = __builtin_amdgcn_wmma_f32_16x16x32_bf16(
                false, pa, false, vb, (short)0, oacc[nt], false, false);
        }
        __builtin_amdgcn_wave_barrier();
    }

    // normalize and write y (bf16, [B,T,H,D] == [8192][768])
    const int b_ = bh / H, h = bh % H;
#pragma unroll
    for (int nt = 0; nt < 4; ++nt) {
        const int d = nt * 16 + ln;
#pragma unroll
        for (int i = 0; i < 8; ++i) {
            const int q = q0 + hi * 8 + i;
            float val = oacc[nt][i] / rsum[i];
            Yb[((size_t)(b_ * T) + q) * C + h * Dh + d] = f2bfbits(val);
        }
    }
}

// ------------------------- output projection GEMM ---------------------------
// Grid: (M_TOT/64, C/64), block 128.
__global__ void __launch_bounds__(128) k_proj(
    const unsigned short* __restrict__ yb,    // [8192][768] bf16
    const unsigned short* __restrict__ WpT,   // [768][768] bf16 (W_proj^T)
    const float* __restrict__ b_proj,         // [768]
    float* __restrict__ out)                  // [8192][768] fp32
{
    const int lane = threadIdx.x & 31;
    const int wave = threadIdx.x >> 5;
    const int hi = lane >> 4, ln = lane & 15;
    const int m0 = blockIdx.x * 64 + wave * 16;
    const int n0 = blockIdx.y * 64;

    v8f acc[4];
#pragma unroll
    for (int nt = 0; nt < 4; ++nt)
#pragma unroll
        for (int i = 0; i < 8; ++i) acc[nt][i] = 0.f;

    const unsigned short* arow = yb + (size_t)(m0 + ln) * C;
    for (int k0 = 0; k0 < C; k0 += 32) {
        __builtin_prefetch(arow + k0 + 32, 0, 1);
        v16bf a;
#pragma unroll
        for (int v = 0; v < 8; ++v) ld_pair(arow + k0 + kmapA(v, hi), a, v * 2);
#pragma unroll
        for (int nt = 0; nt < 4; ++nt) {
            const unsigned short* brow = WpT + (size_t)(n0 + nt * 16 + ln) * C + k0;
            v16bf b;
#pragma unroll
            for (int v = 0; v < 8; ++v) ld_pair(brow + kmapB(v, hi), b, v * 2);
            acc[nt] = __builtin_amdgcn_wmma_f32_16x16x32_bf16(
                false, a, false, b, (short)0, acc[nt], false, false);
        }
    }
#pragma unroll
    for (int nt = 0; nt < 4; ++nt) {
        const int n = n0 + nt * 16 + ln;
        const float bias = b_proj[n];
#pragma unroll
        for (int i = 0; i < 8; ++i) {
            const int m = m0 + hi * 8 + i;
            out[(size_t)m * C + n] = acc[nt][i] + bias;
        }
    }
}

// ------------------------------- launcher -----------------------------------
extern "C" void kernel_launch(void* const* d_in, const int* in_sizes, int n_in,
                              void* d_out, int out_size, void* d_ws, size_t ws_size,
                              hipStream_t stream) {
    (void)in_sizes; (void)n_in; (void)out_size; (void)ws_size;
    const float* x  = (const float*)d_in[0];
    const float* Wa = (const float*)d_in[1];
    const float* ba = (const float*)d_in[2];
    const float* Wp = (const float*)d_in[3];
    const float* bp = (const float*)d_in[4];
    float* out = (float*)d_out;

    char* ws = (char*)d_ws;
    size_t off = 0;
    auto alloc = [&](size_t bytes) -> char* {
        char* p = ws + off;
        off += (bytes + 255) & ~(size_t)255;
        return p;
    };
    unsigned short* xb  = (unsigned short*)alloc((size_t)M_TOT * C * 2);
    unsigned short* WaT = (unsigned short*)alloc((size_t)N_QKV * C * 2);
    unsigned short* WpT = (unsigned short*)alloc((size_t)C * C * 2);
    unsigned short* Qb  = (unsigned short*)alloc((size_t)M_TOT * C * 2);
    unsigned short* Kb  = (unsigned short*)alloc((size_t)M_TOT * C * 2);
    unsigned short* Vt  = (unsigned short*)alloc((size_t)M_TOT * C * 2);
    unsigned short* Yb  = (unsigned short*)alloc((size_t)M_TOT * C * 2);

    const int nx = M_TOT * C;
    k_cvt_bf16<<<(nx + 255) / 256, 256, 0, stream>>>(x, xb, nx);
    const int nwa = C * N_QKV;
    k_transpose_bf16<<<(nwa + 255) / 256, 256, 0, stream>>>(Wa, WaT, C, N_QKV);
    const int nwp = C * C;
    k_transpose_bf16<<<(nwp + 255) / 256, 256, 0, stream>>>(Wp, WpT, C, C);

    k_qkv_rope<<<dim3(M_TOT / 64, N_QKV / 64), 128, 0, stream>>>(
        xb, WaT, ba, Qb, Kb, Vt);
    k_flash<<<dim3(T / 64, Bsz * H), 128, 0, stream>>>(Qb, Kb, Vt, Yb);
    k_proj<<<dim3(M_TOT / 64, C / 64), 128, 0, stream>>>(Yb, WpT, bp, out);
}
